// GFE_10213432230559
// MI455X (gfx1250) — matrix operand order repaired and checked by reading, not compiled, not run on hardware
//
#include <hip/hip_runtime.h>
#include <math.h>

#define HW    64
#define NTOK  4096
#define CH    48
#define CHP   64      // padded channel stride for q/k f16 rows (48..63 = 0)
#define BATCH 2
#define RPE_W 191

typedef _Float16 f16;
typedef __attribute__((ext_vector_type(16))) f16   v16h;
typedef __attribute__((ext_vector_type(8)))  f16   v8h;
typedef __attribute__((ext_vector_type(8)))  float v8f;

__device__ __forceinline__ v8f wmma16(v16h a, v16h b, v8f c) {
  return __builtin_amdgcn_wmma_f32_16x16x32_f16(false, a, false, b, (short)0, c, false, false);
}
__device__ __forceinline__ v16h cat16(v8h lo, v8h hi) {
  return __builtin_shufflevector(lo, hi, 0,1,2,3,4,5,6,7,8,9,10,11,12,13,14,15);
}
__device__ __forceinline__ v8h ld8(const f16* p) { return *(const v8h*)p; }

// ---------------- Kernel A: q = Wq*x + bq ; x2 = BN(inp_w*x + inp_b) ----------------
__global__ void kA(const float* __restrict__ x, const float* __restrict__ Wq, const float* __restrict__ bq,
                   const float* __restrict__ inp_w, const float* __restrict__ inp_b,
                   const float* __restrict__ bn_g, const float* __restrict__ bn_b,
                   const float* __restrict__ bn_mean, const float* __restrict__ bn_var,
                   float* __restrict__ q32, f16* __restrict__ qh, float* __restrict__ x2) {
  __shared__ float sWq[CH*CH];
  __shared__ float sWi[2*CH*CH];
  for (int i = threadIdx.x; i < CH*CH; i += blockDim.x) sWq[i] = Wq[i];
  for (int i = threadIdx.x; i < 2*CH*CH; i += blockDim.x) sWi[i] = inp_w[i];
  __syncthreads();
  int gid = blockIdx.x * blockDim.x + threadIdx.x;
  if (gid >= BATCH*NTOK) return;
  int b = gid / NTOK, n = gid % NTOK;
  const float* xb = x + (size_t)b*CH*NTOK + n;
  float xv[CH];
  #pragma unroll
  for (int c = 0; c < CH; ++c) xv[c] = xb[(size_t)c*NTOK];
  const float qscale = rsqrtf((float)CH);
  f16* qrow = qh + ((size_t)b*NTOK + n)*CHP;
  for (int o = 0; o < CH; ++o) {
    float acc = bq[o];
    #pragma unroll
    for (int c = 0; c < CH; ++c) acc += sWq[o*CH+c]*xv[c];
    q32[(size_t)b*CH*NTOK + (size_t)o*NTOK + n] = acc;
    qrow[o] = (f16)(acc * qscale);
  }
  #pragma unroll
  for (int o = CH; o < CHP; ++o) qrow[o] = (f16)0.f;     // pad
  for (int o = 0; o < 2*CH; ++o) {
    float acc = inp_b[o];
    #pragma unroll
    for (int c = 0; c < CH; ++c) acc += sWi[o*CH+c]*xv[c];
    acc = (acc - bn_mean[o]) * rsqrtf(bn_var[o] + 1e-5f) * bn_g[o] + bn_b[o];
    x2[(size_t)b*2*CH*NTOK + (size_t)o*NTOK + n] = acc;
  }
}

// ---------------- Kernel B: offset head -> pos[B,n,2] ----------------
__global__ void kB(const float* __restrict__ q32, const float* __restrict__ dww, const float* __restrict__ dwb,
                   const float* __restrict__ lng, const float* __restrict__ lnb,
                   const float* __restrict__ pww, float* __restrict__ pos) {
  int gid = blockIdx.x*blockDim.x + threadIdx.x;
  if (gid >= BATCH*NTOK) return;
  int b = gid / NTOK, n = gid % NTOK;
  int y = n >> 6, xq = n & 63;
  float t[CH];
  float mu = 0.f;
  for (int c = 0; c < CH; ++c) {
    const float* qc = q32 + (size_t)b*CH*NTOK + (size_t)c*NTOK;
    const float* wc = dww + c*9;
    float acc = dwb[c];
    #pragma unroll
    for (int ky = 0; ky < 3; ++ky) {
      int yy = y + ky - 1;
      if (yy < 0 || yy >= HW) continue;
      #pragma unroll
      for (int kx = 0; kx < 3; ++kx) {
        int xx = xq + kx - 1;
        if (xx < 0 || xx >= HW) continue;
        acc += qc[yy*HW+xx]*wc[ky*3+kx];
      }
    }
    t[c] = acc; mu += acc;
  }
  mu *= (1.f/CH);
  float var = 0.f;
  #pragma unroll
  for (int c = 0; c < CH; ++c) { float d = t[c]-mu; var += d*d; }
  var *= (1.f/CH);
  float rstd = rsqrtf(var + 1e-5f);
  float offy = 0.f, offx = 0.f;
  #pragma unroll
  for (int c = 0; c < CH; ++c) {
    float v = (t[c]-mu)*rstd*lng[c] + lnb[c];
    v = 0.5f*v*(1.f + erff(v*0.70710678118654752f));   // exact GELU
    offy += pww[c]*v;
    offx += pww[CH+c]*v;
  }
  const float sc = 2.0f/63.0f;                          // orange * OFR
  float py = tanhf(offy)*sc + ((0.5f + (float)y )*(2.f/63.f) - 1.f);
  float px = tanhf(offx)*sc + ((0.5f + (float)xq)*(2.f/63.f) - 1.f);
  pos[((size_t)b*NTOK + n)*2 + 0] = py;
  pos[((size_t)b*NTOK + n)*2 + 1] = px;
}

// ---------------- Kernel C: grid_sample(x, pos) -> k,v (f16) ----------------
__global__ void kC(const float* __restrict__ x, const float* __restrict__ pos,
                   const float* __restrict__ Wk, const float* __restrict__ bk,
                   const float* __restrict__ Wv, const float* __restrict__ bv,
                   f16* __restrict__ kh, f16* __restrict__ vh) {
  __shared__ float sWk[CH*CH], sWv[CH*CH];
  for (int i = threadIdx.x; i < CH*CH; i += blockDim.x) { sWk[i]=Wk[i]; sWv[i]=Wv[i]; }
  __syncthreads();
  int gid = blockIdx.x*blockDim.x + threadIdx.x;
  if (gid >= BATCH*NTOK) return;
  int b = gid/NTOK, n = gid%NTOK;
  float py = pos[((size_t)b*NTOK+n)*2+0];
  float px = pos[((size_t)b*NTOK+n)*2+1];
  float gx = (px+1.f)*0.5f*63.f, gy=(py+1.f)*0.5f*63.f;   // align_corners=True map
  float x0f = floorf(gx), y0f = floorf(gy);
  int x0 = (int)x0f, y0 = (int)y0f;
  float wx1 = gx-x0f, wy1 = gy-y0f;
  int ix[4]; float wt[4];
  #pragma unroll
  for (int c4=0;c4<4;++c4){
    int xc = x0 + (c4&1), yc = y0 + (c4>>1);
    float wgt = ((c4&1)?wx1:1.f-wx1) * ((c4>>1)?wy1:1.f-wy1);
    bool valid = (xc>=0)&&(xc<HW)&&(yc>=0)&&(yc<HW);
    int xcc = min(max(xc,0),HW-1), ycc = min(max(yc,0),HW-1);
    ix[c4] = ycc*HW+xcc; wt[c4] = valid ? wgt : 0.f;
  }
  float xs[CH];
  const float* xb = x + (size_t)b*CH*NTOK;
  #pragma unroll
  for (int c=0;c<CH;++c){
    const float* xc = xb + (size_t)c*NTOK;
    xs[c] = wt[0]*xc[ix[0]] + wt[1]*xc[ix[1]] + wt[2]*xc[ix[2]] + wt[3]*xc[ix[3]];
  }
  f16* krow = kh + ((size_t)b*NTOK + n)*CHP;
  for (int o=0;o<CH;++o){
    float ak=bk[o], av=bv[o];
    #pragma unroll
    for (int c=0;c<CH;++c){ ak += sWk[o*CH+c]*xs[c]; av += sWv[o*CH+c]*xs[c]; }
    krow[o]                                    = (f16)ak;   // [B,n,CHP] for S B-frags
    vh[(size_t)b*CH*NTOK + (size_t)o*NTOK + n] = (f16)av;   // [B,C,n] for PV B-frags
  }
  #pragma unroll
  for (int o=CH;o<CHP;++o) krow[o] = (f16)0.f;              // pad
}

// ---------------- Kernel D: fused deformable flash attention ----------------
__global__ void __launch_bounds__(256) kD(const f16* __restrict__ qh, const f16* __restrict__ kh,
                   const f16* __restrict__ vh, const float* __restrict__ pos,
                   const float* __restrict__ rpe, float* __restrict__ outd) {
  __shared__ f16 sP[8][16*32];                 // per-wave P staging
  const int wv = threadIdx.x >> 5;
  const int lane = threadIdx.x & 31;
  const int hf = lane >> 4, lr = lane & 15;
  const int b  = blockIdx.y;
  const int m0 = blockIdx.x*128 + wv*16;
  const float2* pos2 = (const float2*)pos;
  // Q A-fragments (pre-scaled, padded to 64ch). A-layout => two contiguous 8-half chunks.
  const f16* qrow = qh + ((size_t)b*NTOK + (size_t)(m0 + lr))*CHP;
  v16h aq0 = cat16(ld8(qrow + 8*hf),      ld8(qrow + 16 + 8*hf));
  v16h aq1 = cat16(ld8(qrow + 32 + 8*hf), ld8(qrow + 48 + 8*hf));
  float qgy[8], qgx[8];
  #pragma unroll
  for (int j=0;j<8;++j){
    int mg = m0 + j + 8*hf;
    qgy[j] = (float)(mg>>6)*(2.f/63.f) - 1.f;
    qgx[j] = (float)(mg&63)*(2.f/63.f) - 1.f;
  }
  v8f O0 = {}, O1 = {}, O2 = {};
  float M[8], L[8];
  #pragma unroll
  for (int j=0;j<8;++j){ M[j] = -1e30f; L[j] = 0.f; }
  f16* myP = &sP[wv][0];

  for (int n0 = 0; n0 < NTOK; n0 += 32) {
    __builtin_prefetch(kh + ((size_t)b*NTOK + (size_t)(((n0 + 32) & (NTOK-1)) + lane))*CHP, 0, 1);
    // ---- S = Q*K^T, two 16-col subtiles, K padded to 64
    v8f S[2];
    #pragma unroll
    for (int t=0;t<2;++t){
      const f16* krow = kh + ((size_t)b*NTOK + (size_t)(n0 + 16*t + lr))*CHP;
      v16h bk0 = cat16(ld8(krow + 8*hf),      ld8(krow + 16 + 8*hf));        // K 0..31
      v16h bk1 = cat16(ld8(krow + 32 + 8*hf), ld8(krow + 48 + 8*hf));        // K 32..63
      v8f acc = {};
      acc = wmma16(aq0, bk0, acc);
      acc = wmma16(aq1, bk1, acc);
      S[t] = acc;
    }
    // ---- RPE bias + online softmax (C/D layout: row = j + 8*half, col = lane&15)
    float posy[2], posx[2];
    #pragma unroll
    for (int t=0;t<2;++t){
      float2 p2 = pos2[(size_t)b*NTOK + n0 + 16*t + lr];
      posy[t] = p2.x; posx[t] = p2.y;
    }
    float corrf[8];
    #pragma unroll
    for (int j=0;j<8;++j){
      float sv[2];
      #pragma unroll
      for (int t=0;t<2;++t){
        float dy = (qgy[j]-posy[t])*0.5f;
        float dx = (qgx[j]-posx[t])*0.5f;
        float gx = (dx+1.f)*0.5f*(float)(RPE_W-1);
        float gy = (dy+1.f)*0.5f*(float)(RPE_W-1);
        float x0f = floorf(gx), y0f = floorf(gy);
        int x0=(int)x0f, y0=(int)y0f;
        float wx1 = gx-x0f, wy1=gy-y0f;
        float bias = 0.f;
        #pragma unroll
        for (int c4=0;c4<4;++c4){
          int xc = x0 + (c4&1), yc = y0 + (c4>>1);
          float wgt = ((c4&1)?wx1:1.f-wx1) * ((c4>>1)?wy1:1.f-wy1);
          bool valid = (xc>=0)&&(xc<RPE_W)&&(yc>=0)&&(yc<RPE_W);
          int xcc = min(max(xc,0),RPE_W-1), ycc = min(max(yc,0),RPE_W-1);
          float wz = valid ? wgt : 0.f;                 // zero weight, unconditional load
          bias += wz * rpe[ycc*RPE_W+xcc];
        }
        sv[t] = S[t][j] + bias;
      }
      float mx = fmaxf(sv[0], sv[1]);
      #pragma unroll
      for (int off=1; off<16; off<<=1) mx = fmaxf(mx, __shfl_xor(mx, off, 16));
      float Mn = fmaxf(M[j], mx);
      float corr = __expf(M[j]-Mn);
      float p0 = __expf(sv[0]-Mn), p1 = __expf(sv[1]-Mn);
      float rs = p0+p1;
      #pragma unroll
      for (int off=1; off<16; off<<=1) rs += __shfl_xor(rs, off, 16);
      L[j] = L[j]*corr + rs;
      M[j] = Mn;
      corrf[j] = corr;
      int row = j + 8*hf;
      myP[row*32 + lr]      = (f16)p0;
      myP[row*32 + 16 + lr] = (f16)p1;
    }
    #pragma unroll
    for (int j=0;j<8;++j){ O0[j]*=corrf[j]; O1[j]*=corrf[j]; O2[j]*=corrf[j]; }
    // ---- reload P as A-fragment (per-wave LDS region, DS in-order within wave)
    const f16* prow = myP + lr*32;
    v16h aP = cat16(ld8(prow + 8*hf), ld8(prow + 16 + 8*hf));
    // ---- O += P * V^T  (V stored [B,C,n]: each lane reads two 8-half chunks, 16 halves apart)
    #pragma unroll
    for (int ct=0;ct<3;++ct){
      const f16* vrow = vh + (size_t)b*CH*NTOK + (size_t)(16*ct+lr)*NTOK + n0 + 8*hf;
      v16h bV = cat16(ld8(vrow), ld8(vrow + 16));
      if (ct==0)      O0 = wmma16(aP, bV, O0);
      else if (ct==1) O1 = wmma16(aP, bV, O1);
      else            O2 = wmma16(aP, bV, O2);
    }
  }
  #pragma unroll
  for (int j=0;j<8;++j){
    float inv = 1.f / L[j];
    int mg = m0 + j + 8*hf;
    size_t base = (size_t)b*CH*NTOK + mg;
    outd[base + (size_t)(lr)   *NTOK] = O0[j]*inv;
    outd[base + (size_t)(16+lr)*NTOK] = O1[j]*inv;
    outd[base + (size_t)(32+lr)*NTOK] = O2[j]*inv;
  }
}

// ---------------- Kernel E: window attention (3 splits, c=16, 64 tokens) ----------------
__global__ void __launch_bounds__(128) kE(const float* __restrict__ x2, float* __restrict__ wo) {
  __shared__ f16 sQ[64][32];      // rows padded to 32, halves 16..31 = 0
  __shared__ f16 sVt[16][72];     // transposed V, padded stride (16B aligned rows)
  __shared__ f16 sPw[4][16*64];
  const int split = blockIdx.z;
  const int b = blockIdx.y;
  const int win = blockIdx.x;
  int dh, dw;
  if (split==0){dh=8;dw=8;} else if (split==1){dh=64;dw=1;} else {dh=1;dw=64;}
  const int nwx = HW/dw;
  const int wy = win / nwx, wx = win % nwx;
  for (int i = threadIdx.x; i < 64*16; i += 128){
    int tk = i >> 4, c = i & 15;
    int iy = tk / dw, ixn = tk % dw;
    int n = (wy*dh + iy)*HW + wx*dw + ixn;
    sQ[tk][c]      = (f16)x2[(size_t)b*2*CH*NTOK + (size_t)(split*32 + c)*NTOK + n];
    sQ[tk][16 + c] = (f16)0.f;
    sVt[c][tk]     = (f16)x2[(size_t)b*2*CH*NTOK + (size_t)(split*32 + 16 + c)*NTOK + n];
  }
  __syncthreads();
  const int wvi = threadIdx.x >> 5;
  const int lane = threadIdx.x & 31;
  const int hf = lane>>4, lr = lane&15;
  const int mr0 = wvi*16;
  const f16* qr = &sQ[mr0+lr][0];
  v16h aQ = cat16(ld8(qr + 8*hf), ld8(qr + 16 + 8*hf));   // K 16..31 read zeros
  v8f S[4];
  #pragma unroll
  for (int nt=0;nt<4;++nt){
    const f16* br = &sQ[nt*16+lr][0];
    v16h bQ = cat16(ld8(br + 8*hf), ld8(br + 16 + 8*hf)); // K 16..31 read zeros
    v8f acc = {};
    S[nt] = wmma16(aQ, bQ, acc);
  }
  f16* myP = &sPw[wvi][0];
  float Lr[8];
  #pragma unroll
  for (int j=0;j<8;++j){
    float mx = fmaxf(fmaxf(S[0][j],S[1][j]), fmaxf(S[2][j],S[3][j]));
    #pragma unroll
    for (int off=1; off<16; off<<=1) mx = fmaxf(mx, __shfl_xor(mx, off, 16));
    float p[4], rs = 0.f;
    #pragma unroll
    for (int nt=0;nt<4;++nt){ p[nt] = __expf(S[nt][j]-mx); rs += p[nt]; }
    #pragma unroll
    for (int off=1; off<16; off<<=1) rs += __shfl_xor(rs, off, 16);
    Lr[j] = rs;
    int row = j + 8*hf;
    #pragma unroll
    for (int nt=0;nt<4;++nt) myP[row*64 + nt*16 + lr] = (f16)p[nt];
  }
  v8f O = {};
  #pragma unroll
  for (int kt=0;kt<2;++kt){
    const f16* prow = myP + lr*64 + 32*kt;
    v16h aP = cat16(ld8(prow + 8*hf), ld8(prow + 16 + 8*hf));
    const f16* vr = &sVt[lr][32*kt + 8*hf];
    v16h bV = cat16(ld8(vr), ld8(vr + 16));
    O = wmma16(aP, bV, O);
  }
  #pragma unroll
  for (int j=0;j<8;++j){
    int tk = mr0 + j + 8*hf;
    int iy = tk / dw, ixn = tk % dw;
    int n = (wy*dh+iy)*HW + wx*dw + ixn;
    wo[(size_t)b*CH*NTOK + (size_t)(split*16+lr)*NTOK + n] = O[j] / Lr[j];
  }
}

// ---------------- Kernel F: pout + blend ----------------
__global__ void kF(const float* __restrict__ wo, const float* __restrict__ pw, const float* __restrict__ pb,
                   const float* __restrict__ outd, float* __restrict__ out) {
  __shared__ float sW[CH*CH];
  for (int i=threadIdx.x;i<CH*CH;i+=blockDim.x) sW[i]=pw[i];
  __syncthreads();
  int gid = blockIdx.x*blockDim.x + threadIdx.x;
  if (gid >= BATCH*NTOK) return;
  int b = gid/NTOK, n = gid%NTOK;
  float wv[CH];
  #pragma unroll
  for (int c=0;c<CH;++c) wv[c] = wo[(size_t)b*CH*NTOK + (size_t)c*NTOK + n];
  for (int o=0;o<CH;++o){
    float acc = pb[o];
    #pragma unroll
    for (int c=0;c<CH;++c) acc += sW[o*CH+c]*wv[c];
    size_t idx = (size_t)b*CH*NTOK + (size_t)o*NTOK + n;
    out[idx] = 0.5f*acc + 0.5f*outd[idx];
  }
}

extern "C" void kernel_launch(void* const* d_in, const int* in_sizes, int n_in,
                              void* d_out, int out_size, void* d_ws, size_t ws_size,
                              hipStream_t stream) {
  const float* x      = (const float*)d_in[0];
  const float* Wq     = (const float*)d_in[1];
  const float* bq     = (const float*)d_in[2];
  const float* dww    = (const float*)d_in[3];
  const float* dwb    = (const float*)d_in[4];
  const float* lng    = (const float*)d_in[5];
  const float* lnb    = (const float*)d_in[6];
  const float* pww    = (const float*)d_in[7];
  const float* Wk     = (const float*)d_in[8];
  const float* bk     = (const float*)d_in[9];
  const float* Wv     = (const float*)d_in[10];
  const float* bv     = (const float*)d_in[11];
  const float* inp_w  = (const float*)d_in[12];
  const float* inp_b  = (const float*)d_in[13];
  const float* bn_g   = (const float*)d_in[14];
  const float* bn_b   = (const float*)d_in[15];
  const float* bn_m   = (const float*)d_in[16];
  const float* bn_v   = (const float*)d_in[17];
  const float* pout_w = (const float*)d_in[18];
  const float* pout_b = (const float*)d_in[19];
  const float* rpe    = (const float*)d_in[20];

  char* ws = (char*)d_ws;
  size_t off = 0;
  auto alloc = [&](size_t bytes) { char* p = ws + off; off = (off + bytes + 255) & ~(size_t)255; return p; };
  float* q32  = (float*)alloc((size_t)BATCH*CH*NTOK*4);
  f16*   qh   = (f16*)  alloc((size_t)BATCH*NTOK*CHP*2);
  f16*   kh   = (f16*)  alloc((size_t)BATCH*NTOK*CHP*2);
  f16*   vh   = (f16*)  alloc((size_t)BATCH*CH*NTOK*2);
  float* x2   = (float*)alloc((size_t)BATCH*2*CH*NTOK*4);
  float* pos  = (float*)alloc((size_t)BATCH*NTOK*2*4);
  float* outd = (float*)alloc((size_t)BATCH*CH*NTOK*4);
  float* wo   = (float*)alloc((size_t)BATCH*CH*NTOK*4);

  const int total = BATCH*NTOK;        // 8192
  kA<<<(total+255)/256, 256, 0, stream>>>(x, Wq, bq, inp_w, inp_b, bn_g, bn_b, bn_m, bn_v, q32, qh, x2);
  kB<<<(total+255)/256, 256, 0, stream>>>(q32, dww, dwb, lng, lnb, pww, pos);
  kC<<<(total+255)/256, 256, 0, stream>>>(x, pos, Wk, bk, Wv, bv, kh, vh);
  kD<<<dim3(NTOK/128, BATCH), 256, 0, stream>>>(qh, kh, vh, pos, rpe, outd);
  kE<<<dim3(64, BATCH, 3), 128, 0, stream>>>(x2, wo);
  kF<<<(total+255)/256, 256, 0, stream>>>(wo, pout_w, pout_b, outd, (float*)d_out);
}
